// MTCNN_26319559590160
// MI455X (gfx1250) — compile-verified, matrix operand order checked by reading, not run on hardware
//
#include <hip/hip_runtime.h>
#include <hip/hip_bf16.h>

typedef __attribute__((ext_vector_type(16))) _Float16     v16h;
typedef __attribute__((ext_vector_type(8)))  _Float16     v8h;
typedef __attribute__((ext_vector_type(8)))  float        v8f;
typedef __attribute__((ext_vector_type(4)))  unsigned int u32x4;
typedef __attribute__((ext_vector_type(4)))  int          i32x4;
typedef __attribute__((ext_vector_type(8)))  int          i32x8;

#define NEGV (-1e30f)

// ---------------------------------------------------------------------------
// Geometry: 513 -> conv1(3x3) -> 511 -> pool(ceil2) -> 256 -> conv2 -> 254
//           -> conv3 -> 252.   Hp=Wp=252, Hp*Wp = 63504.
// ---------------------------------------------------------------------------
#define BATCH   8
#define HP      252
#define NPIX    (HP*HP)          // 63504
#define KPAD    160              // im2col K: k = tap*16 + c, 9*16=144 -> 160

// ---------------------------------------------------------------------------
// Tensor Data Mover: 1-D tile copy global -> LDS (weight panel staging).
// D# packing per CDNA5 ISA ch.8 (group0 128b, group1 256b, groups 2/3 zero).
// Guarded: falls back to a cooperative copy when the builtin is absent.
// ---------------------------------------------------------------------------
#if defined(__gfx1250__) && __has_builtin(__builtin_amdgcn_tensor_load_to_lds) && \
    __has_builtin(__builtin_amdgcn_s_wait_tensorcnt)
#define HAVE_TDM 1
#endif

#ifdef HAVE_TDM
__device__ __forceinline__ void tdm_copy_to_lds(const void* gsrc,
                                                unsigned lds_off_bytes,
                                                unsigned nelem_u16) {
  unsigned long long ga = (unsigned long long)gsrc;
  u32x4 g0;
  g0[0] = 1u;                                   // count=1, user mode
  g0[1] = lds_off_bytes;                        // lds_addr (bytes)
  g0[2] = (unsigned)(ga & 0xffffffffu);         // global_addr[31:0]
  g0[3] = (unsigned)((ga >> 32) & 0x01ffffffu)  // global_addr[56:32]
          | 0x80000000u;                        // type = 2 ("image")
  i32x8 g1;
  g1[0] = (int)(1u << 16);                      // data_size=1 -> 2 bytes
  g1[1] = (int)((nelem_u16 & 0xffffu) << 16);   // tensor_dim0[15:0]
  g1[2] = (int)((nelem_u16 >> 16) & 0xffffu);   // tensor_dim0[31:16]
  g1[3] = (int)((nelem_u16 & 0xffffu) << 16);   // tile_dim0 = nelem
  g1[4] = 0;                                    // tile_dim1/2 = 0 (unused)
  g1[5] = (int)nelem_u16;                       // tensor_dim0_stride
  g1[6] = 0;
  g1[7] = 0;
  i32x4 z4 = {0, 0, 0, 0};
#if __clang_major__ >= 23
  i32x8 z8 = {0, 0, 0, 0, 0, 0, 0, 0};
  __builtin_amdgcn_tensor_load_to_lds(g0, g1, z4, z4, z8, 0);
#else
  __builtin_amdgcn_tensor_load_to_lds(g0, g1, z4, z4, 0);
#endif
}
#endif

// ---------------------------------------------------------------------------
// Kernel 1: weights -> f16, im2col-K layout k = tap*16 + c (c padded to 16),
// K padded to 160 (5 WMMA chunks of 32 = 2 taps x 16 ch each).
// ---------------------------------------------------------------------------
__global__ void prep_weights(const float* __restrict__ w2,
                             const float* __restrict__ w3,
                             _Float16* __restrict__ w2h,
                             _Float16* __restrict__ w3h) {
  int t = blockIdx.x * blockDim.x + threadIdx.x;
  int stride = gridDim.x * blockDim.x;
  for (int i = t; i < 16 * KPAD; i += stride) {
    int m = i / KPAD, k = i % KPAD;
    int tap = k >> 4, c = k & 15;
    float v = (k < 144 && c < 10) ? w2[m * 90 + c * 9 + tap] : 0.0f;
    w2h[i] = (_Float16)v;
  }
  for (int i = t; i < 32 * KPAD; i += stride) {
    int m = i / KPAD, k = i % KPAD;
    int tap = k >> 4, c = k & 15;
    float v = (k < 144) ? w3[m * 144 + c * 9 + tap] : 0.0f;
    w3h[i] = (_Float16)v;
  }
}

// ---------------------------------------------------------------------------
// Kernel 2: fused conv1(3->10, 3x3, VALID) + PReLU + 2x2 ceil-maxpool.
// K=27 too skinny for matrix cores -> fp32 VALU, one thread per pooled
// output, shared 3x4x4 input patch for the 4 pooled conv taps.
// ---------------------------------------------------------------------------
__global__ void conv1_pool(const float* __restrict__ img,
                           const float* __restrict__ w1,
                           const float* __restrict__ b1,
                           const float* __restrict__ a1,
                           _Float16* __restrict__ pool1) {
  int t = blockIdx.x * blockDim.x + threadIdx.x;
  const int total = BATCH * 10 * 256 * 256;
  if (t >= total) return;
  int px = t & 255;
  int py = (t >> 8) & 255;
  int c  = (t >> 16) % 10;
  int b  = t / (10 * 65536);

  float w[27];
#pragma unroll
  for (int i = 0; i < 27; ++i) w[i] = w1[c * 27 + i];

  float in[3][4][4];
#pragma unroll
  for (int ic = 0; ic < 3; ++ic)
#pragma unroll
    for (int i = 0; i < 4; ++i) {
      int row = 2 * py + i; if (row > 512) row = 512;
#pragma unroll
      for (int j = 0; j < 4; ++j) {
        int col = 2 * px + j; if (col > 512) col = 512;
        in[ic][i][j] =
            (img[((size_t)(b * 3 + ic) * 513 + row) * 513 + col] - 127.5f) *
            0.0078125f;
      }
    }

  float aa = a1[c], bb = b1[c];
  float best = -INFINITY;
#pragma unroll
  for (int dy = 0; dy < 2; ++dy)
#pragma unroll
    for (int dx = 0; dx < 2; ++dx) {
      if (2 * py + dy >= 511 || 2 * px + dx >= 511) continue;
      float s = bb;
#pragma unroll
      for (int ic = 0; ic < 3; ++ic)
#pragma unroll
        for (int ty = 0; ty < 3; ++ty)
#pragma unroll
          for (int tx = 0; tx < 3; ++tx)
            s += in[ic][dy + ty][dx + tx] * w[(ic * 3 + ty) * 3 + tx];
      s = s > 0.0f ? s : aa * s;
      best = fmaxf(best, s);
    }
  pool1[((size_t)(b * 10 + c) * 256 + py) * 256 + px] = (_Float16)best;
}

// ---------------------------------------------------------------------------
// Templated WMMA conv (3x3 VALID, CIN -> MTILES*16 ch) + PReLU.
//
// Block = 256 threads = 8 waves = MTILES m-tiles x NSUB x-subtiles over a
// strip of P = NSUB*16 output pixels of one row.
//
// Phase 0: TDM (tensor_load_to_lds) stages the weight panel into LDS.
// Phase 1: branchless cooperative im2col into LDS [N][KPAD], k = tap*16+c
//          (shift/mask address math only; rep masked so channel predicates
//          fold at compile time) + global_prefetch of the next strip.
// Phase 2: per the CDNA5 16-bit fragment layout (VGPR v <-> K-pair
//          (v&3)*2 + (v>>2)*16 + h*8), each lane's 16 halves per K-chunk are
//          two contiguous 16-byte runs -> 2x ds_load_b128 (A and B) per
//          chunk, then one v_wmma_f32_16x16x32_f16.  5 chunks cover K=160.
// ---------------------------------------------------------------------------
template <int CIN, int MTILES, int NSUB, int WIN, int WOUT>
__global__ __launch_bounds__(256) void conv_wmma(
    const _Float16* __restrict__ in,    // [B][CIN][WIN][WIN]
    const _Float16* __restrict__ wgt,   // [MTILES*16][KPAD]
    const float* __restrict__ bias,
    const float* __restrict__ alpha,
    _Float16* __restrict__ out) {       // [B][MTILES*16][WOUT][WOUT]
  constexpr int P    = NSUB * 16;       // strip width in pixels
  constexpr int REPS = 256 / P;         // threads per pixel in the fill
  constexpr int W    = MTILES * 16;     // output channels

  __shared__ _Float16 lds_b[P][KPAD];
  __shared__ _Float16 lds_w[W][KPAD];

  const int tid = threadIdx.x;
  const int xb  = blockIdx.x * P;
  const int y   = blockIdx.y;           // 0..WOUT-1
  const int b   = blockIdx.z;

  // ---- Phase 0: weight panel -> LDS ----
#ifdef HAVE_TDM
  if (tid < 32) {                       // one wave issues the DMA
    tdm_copy_to_lds(wgt, (unsigned)(size_t)&lds_w[0][0], (unsigned)(W * KPAD));
    __builtin_amdgcn_s_wait_tensorcnt(0);
  }
#else
  {
    const unsigned* src = (const unsigned*)wgt;
    unsigned* dst = (unsigned*)&lds_w[0][0];
    for (int i = tid; i < W * KPAD / 2; i += 256) dst[i] = src[i];
  }
#endif

  // ---- Phase 1: branchless im2col fill ----
  {
    const int n   = tid & (P - 1);
    const int rep = (tid / P) & (REPS - 1);   // bounded -> predicates fold
    const int xin = xb + n;
    // speculative prefetch of the next strip of the middle input row
    __builtin_prefetch(&in[(size_t)(b * CIN) * WIN * WIN +
                           (size_t)(y + 1) * WIN + xin + P], 0, 1);
#pragma unroll
    for (int tap = 0; tap < 9; ++tap) {
      const int ty = tap / 3, tx = tap % 3;
      int xx = xin + tx; if (xx > WIN - 1) xx = WIN - 1;
      const size_t rowoff = (size_t)(y + ty) * WIN + xx;
#pragma unroll
      for (int c = 0; c < 16; c += REPS) {
        const int cc = c + rep;               // range provably [c, c+REPS-1]
        _Float16 v = (_Float16)0.0f;
        if (cc < CIN)                          // compile-time constant now
          v = in[(size_t)(b * CIN + cc) * WIN * WIN + rowoff];
        lds_b[n][tap * 16 + cc] = v;
      }
    }
#pragma unroll
    for (int c = 0; c < 16; c += REPS)         // K pad 144..159
      lds_b[n][144 + c + rep] = (_Float16)0.0f;
  }
  __syncthreads();

  // ---- Phase 2: WMMA ----
  const int wave  = (tid >> 5) & 7;
  const int lane  = tid & 31;
  const int h     = lane >> 4;
  const int nidx  = lane & 15;
  const int sub   = wave & (NSUB - 1);
  const int mbase = (wave / NSUB) * 16;

  const _Float16* arow = &lds_w[mbase + nidx][0];
  const _Float16* brow = &lds_b[sub * 16 + nidx][0];

  v8f acc = {};
#pragma unroll
  for (int kc = 0; kc < 5; ++kc) {
    const int o1 = kc * 32 + h * 8;      // halves; 16B-aligned byte offsets
    const int o2 = o1 + 16;
    v8h a_lo = *(const v8h*)(arow + o1);
    v8h a_hi = *(const v8h*)(arow + o2);
    v8h b_lo = *(const v8h*)(brow + o1);
    v8h b_hi = *(const v8h*)(brow + o2);
    v16h A = __builtin_shufflevector(a_lo, a_hi, 0, 1, 2, 3, 4, 5, 6, 7,
                                     8, 9, 10, 11, 12, 13, 14, 15);
    v16h B = __builtin_shufflevector(b_lo, b_hi, 0, 1, 2, 3, 4, 5, 6, 7,
                                     8, 9, 10, 11, 12, 13, 14, 15);
    acc = __builtin_amdgcn_wmma_f32_16x16x32_f16(false, A, false, B,
                                                 (short)0, acc, false, false);
  }

  // ---- store: C/D layout: VGPR r -> M = r + h*8, N = lane&15 ----
  const int x = xb + sub * 16 + nidx;
  if (x < WOUT) {
#pragma unroll
    for (int r = 0; r < 8; ++r) {
      int m = mbase + r + h * 8;
      float v = acc[r] + bias[m];
      v = v > 0.0f ? v : alpha[m] * v;
      out[((size_t)(b * W + m) * WOUT + y) * WOUT + x] = (_Float16)v;
    }
  }
}

// ---------------------------------------------------------------------------
// Kernel 5: 1x1 heads: softmax prob, 4-ch regression, grid box coords.
// ---------------------------------------------------------------------------
__global__ void heads_kernel(const _Float16* __restrict__ conv3buf,
                             const float* __restrict__ w41,
                             const float* __restrict__ b41,
                             const float* __restrict__ w42,
                             const float* __restrict__ b42,
                             float* __restrict__ out,
                             float* __restrict__ s0) {
  int t = blockIdx.x * blockDim.x + threadIdx.x;
  const int total = BATCH * NPIX;
  if (t >= total) return;
  int b = t / NPIX;
  int j = t % NPIX;
  int y = j / HP, x = j % HP;

  float v[32];
#pragma unroll
  for (int c = 0; c < 32; ++c)
    v[c] = (float)conv3buf[((size_t)(b * 32 + c) * HP + y) * HP + x];

  float z0 = b41[0], z1 = b41[1];
#pragma unroll
  for (int c = 0; c < 32; ++c) { z0 += w41[c] * v[c]; z1 += w41[32 + c] * v[c]; }
  float mz = fmaxf(z0, z1);
  float e0 = __expf(z0 - mz), e1 = __expf(z1 - mz);
  float p1 = e1 / (e0 + e1);

  float rg[4];
#pragma unroll
  for (int q = 0; q < 4; ++q) {
    float s = b42[q];
#pragma unroll
    for (int c = 0; c < 32; ++c) s += w42[q * 32 + c] * v[c];
    rg[q] = s;
  }

  float fx = (float)x, fy = (float)y;
  float q1x = floorf((2.0f * fx + 1.0f)  / 0.25f);
  float q1y = floorf((2.0f * fy + 1.0f)  / 0.25f);
  float q2x = floorf((2.0f * fx + 12.0f) / 0.25f);
  float q2y = floorf((2.0f * fy + 12.0f) / 0.25f);

  size_t base = ((size_t)b * NPIX + j) * 9;
  out[base + 0] = q1x; out[base + 1] = q1y;
  out[base + 2] = q2x; out[base + 3] = q2y;
  out[base + 4] = p1;
  out[base + 5] = rg[0]; out[base + 6] = rg[1];
  out[base + 7] = rg[2]; out[base + 8] = rg[3];

  s0[(size_t)b * NPIX + j] = (p1 >= 0.6f) ? p1 : NEGV;
}

// ---------------------------------------------------------------------------
// Kernel 6: greedy NMS, one 1024-thread workgroup per image: 100 rounds of
// LDS-tree argmax (first-index tiebreak) + IoU suppression, then zero
// non-kept 9-float rows of d_out.
// ---------------------------------------------------------------------------
__global__ void nms_kernel(float* __restrict__ out,
                           float* __restrict__ s,
                           int* __restrict__ keep) {
  const int n = NPIX;
  const int b = blockIdx.x;
  const int tid = threadIdx.x;
  float* sb = s + (size_t)b * n;
  int*   kb = keep + (size_t)b * n;
  float* ob = out + (size_t)b * n * 9;

  __shared__ float sv[1024];
  __shared__ int   si[1024];

  for (int j = tid; j < n; j += 1024) kb[j] = 0;
  __syncthreads();

  for (int it = 0; it < 100; ++it) {
    float bv = -INFINITY; int bi = 0x7fffffff;
    for (int j = tid; j < n; j += 1024) {
      float vv = sb[j];
      if (vv > bv) { bv = vv; bi = j; }
    }
    sv[tid] = bv; si[tid] = bi;
    __syncthreads();
    for (int off = 512; off > 0; off >>= 1) {
      if (tid < off) {
        float v2 = sv[tid + off]; int i2 = si[tid + off];
        if (v2 > sv[tid] || (v2 == sv[tid] && i2 < si[tid])) {
          sv[tid] = v2; si[tid] = i2;
        }
      }
      __syncthreads();
    }
    float sbest = sv[0];
    int ib = si[0];
    __syncthreads();
    if (!(sbest > NEGV)) break;

    float x1 = ob[(size_t)ib * 9 + 0], y1 = ob[(size_t)ib * 9 + 1];
    float x2 = ob[(size_t)ib * 9 + 2], y2 = ob[(size_t)ib * 9 + 3];
    float areai = (x2 - x1 + 1.0f) * (y2 - y1 + 1.0f);
    for (int j = tid; j < n; j += 1024) {
      float xj1 = ob[(size_t)j * 9 + 0], yj1 = ob[(size_t)j * 9 + 1];
      float xj2 = ob[(size_t)j * 9 + 2], yj2 = ob[(size_t)j * 9 + 3];
      float xx1 = fmaxf(x1, xj1), yy1 = fmaxf(y1, yj1);
      float xx2 = fminf(x2, xj2), yy2 = fminf(y2, yj2);
      float inter = fmaxf(0.0f, xx2 - xx1 + 1.0f) *
                    fmaxf(0.0f, yy2 - yy1 + 1.0f);
      float areaj = (xj2 - xj1 + 1.0f) * (yj2 - yj1 + 1.0f);
      float iou = inter / (areai + areaj - inter);
      if (iou > 0.5f) sb[j] = NEGV;
    }
    if (tid == 0) kb[ib] = 1;
    __syncthreads();
  }

  __syncthreads();
  for (int j = tid; j < n; j += 1024) {
    if (!kb[j]) {
#pragma unroll
      for (int q = 0; q < 9; ++q) ob[(size_t)j * 9 + q] = 0.0f;
    }
  }
}

// ---------------------------------------------------------------------------
// Workspace layout (bytes, all 256-aligned)
// ---------------------------------------------------------------------------
#define O_W2H   ((size_t)0)            // 16*160*2  = 5120
#define O_W3H   ((size_t)5120)         // 32*160*2  = 10240
#define O_POOL1 ((size_t)15360)        // 8*10*256*256*2 = 10485760
#define O_CONV2 ((size_t)10501120)     // 8*16*254*254*2 = 16516096
#define O_CONV3 ((size_t)27017216)     // 8*32*252*252*2 = 32514048
#define O_S     ((size_t)59531264)     // 8*63504*4 = 2032128
#define O_KEEP  ((size_t)61563392)     // 8*63504*4 = 2032128
// total ~63.6 MB

extern "C" void kernel_launch(void* const* d_in, const int* in_sizes, int n_in,
                              void* d_out, int out_size, void* d_ws, size_t ws_size,
                              hipStream_t stream) {
  const float* imgs = (const float*)d_in[0];
  const float* w1   = (const float*)d_in[1];
  const float* b1   = (const float*)d_in[2];
  const float* a1   = (const float*)d_in[3];
  const float* w2   = (const float*)d_in[4];
  const float* b2   = (const float*)d_in[5];
  const float* a2   = (const float*)d_in[6];
  const float* w3   = (const float*)d_in[7];
  const float* b3   = (const float*)d_in[8];
  const float* a3   = (const float*)d_in[9];
  const float* w41  = (const float*)d_in[10];
  const float* b41  = (const float*)d_in[11];
  const float* w42  = (const float*)d_in[12];
  const float* b42  = (const float*)d_in[13];
  float* out = (float*)d_out;

  char* ws = (char*)d_ws;
  _Float16* w2h   = (_Float16*)(ws + O_W2H);
  _Float16* w3h   = (_Float16*)(ws + O_W3H);
  _Float16* pool1 = (_Float16*)(ws + O_POOL1);
  _Float16* c2buf = (_Float16*)(ws + O_CONV2);
  _Float16* c3buf = (_Float16*)(ws + O_CONV3);
  float*    sbuf  = (float*)(ws + O_S);
  int*      kbuf  = (int*)(ws + O_KEEP);

  // 1) weight prep (f32 -> padded-K f16, k = tap*16 + c)
  prep_weights<<<4, 256, 0, stream>>>(w2, w3, w2h, w3h);

  // 2) fused conv1 + PReLU + ceil-maxpool
  {
    int total = BATCH * 10 * 256 * 256;
    conv1_pool<<<(total + 255) / 256, 256, 0, stream>>>(imgs, w1, b1, a1, pool1);
  }

  // 3) conv2: 10->16ch, 256^2 -> 254^2. 1 m-tile x 8 x-subtiles (128-px strip)
  conv_wmma<10, 1, 8, 256, 254>
      <<<dim3(2, 254, BATCH), 256, 0, stream>>>(pool1, w2h, b2, a2, c2buf);

  // 4) conv3: 16->32ch, 254^2 -> 252^2. 2 m-tiles x 4 x-subtiles (64-px strip)
  conv_wmma<16, 2, 4, 254, 252>
      <<<dim3(4, 252, BATCH), 256, 0, stream>>>(c2buf, w3h, b3, a3, c3buf);

  // 5) heads: softmax + reg + grid boxes -> d_out, thresholded scores -> ws
  {
    int total = BATCH * NPIX;
    heads_kernel<<<(total + 255) / 256, 256, 0, stream>>>(c3buf, w41, b41, w42,
                                                          b42, out, sbuf);
  }

  // 6) per-image NMS + row masking
  nms_kernel<<<BATCH, 1024, 0, stream>>>(out, sbuf, kbuf);
}
